// FlexScaleSeqAttention_84207128805783
// MI455X (gfx1250) — matrix operand order repaired and checked by reading, not compiled, not run on hardware
//
#include <hip/hip_runtime.h>
#include <cstdint>
#include <cstddef>

// ---------------------------------------------------------------------------
// FlexScaleSeqAttention for MI455X (gfx1250, wave32, WMMA bf16 16x16x32,
// async global->LDS staging with ASYNCcnt double buffering).
//
// Algebraic restructure:
//   out = wf @ (attn^T @ v) + bf * colsum(v)      (per b,h)
// reduces 610 GFLOP -> ~60 GFLOP and avoids the 536MB attn/flex tensors.
// ---------------------------------------------------------------------------

typedef __bf16 bf16;
typedef __attribute__((ext_vector_type(16))) __bf16 v16bf;
typedef __attribute__((ext_vector_type(8)))  float  v8f;

#define S_LEN 2048
#define CIN   1024
#define NH    16
#define HD    64
#define BATCH 2
#define BHN   (BATCH*NH)

__device__ __forceinline__ bf16 f2bf(float x) {
  union { float f; uint32_t u; } a; a.f = x;
  uint32_t r = a.u + 0x7FFFu + ((a.u >> 16) & 1u);   // round-to-nearest-even
  uint16_t h = (uint16_t)(r >> 16);
  bf16 o; __builtin_memcpy(&o, &h, 2); return o;
}
__device__ __forceinline__ float bf2f(bf16 x) {
  uint16_t h; __builtin_memcpy(&h, &x, 2);
  union { uint32_t u; float f; } a; a.u = ((uint32_t)h) << 16; return a.f;
}

// Async DMA: 16B global -> LDS, tracked by ASYNCcnt (gfx1250 VGLOBAL op 98).
// VDST holds the per-lane LDS byte offset; 64-bit vaddr form ("off" saddr).
__device__ __forceinline__ void async_ld_b128(const bf16* lds_ptr, const bf16* gptr) {
  uint32_t lds_off = (uint32_t)(uintptr_t)lds_ptr;   // low 32 bits of LDS aperture addr
  asm volatile("global_load_async_to_lds_b128 %0, %1, off"
               :: "v"(lds_off), "v"(gptr)
               : "memory");
}
// Async loads retire in order, so waiting cnt<=N (N = loads just issued for the
// NEXT tile) guarantees the CURRENT tile is fully resident in LDS.
#define WAIT_ASYNC(n) asm volatile("s_wait_asynccnt " #n ::: "memory")

union FragU { v16bf v; uint4 u[2]; };

// A-matrix fragment (16x32 bf16): lane m = row, element e -> k = 16*(e>>3)+8*hi+(e&7)
__device__ __forceinline__ v16bf frag_a(const bf16* lds, int row, int stride, int k0, int hi) {
  FragU f;
  const bf16* b = lds + row * stride + k0;
  f.u[0] = *(const uint4*)(b + 8 * hi);
  f.u[1] = *(const uint4*)(b + 16 + 8 * hi);
  return f.v;
}
// B-matrix fragment (32x16 bf16): lane n = row of LDS [n][k]; element e -> k = 16*(e>>3)+8*hi+(e&7)
__device__ __forceinline__ v16bf frag_b(const bf16* lds, int row, int stride, int k0, int hi) {
  FragU f;
  const bf16* b = lds + row * stride + k0;
  f.u[0] = *(const uint4*)(b + 8 * hi);
  f.u[1] = *(const uint4*)(b + 16 + 8 * hi);
  return f.v;
}
__device__ __forceinline__ v8f wmma_bf16(v16bf a, v16bf b, v8f c) {
  return __builtin_amdgcn_wmma_f32_16x16x32_bf16(false, a, false, b, (short)0, c, false, false);
}

// ---------------------------------------------------------------------------
// K0: fp32 -> bf16 conversion
// ---------------------------------------------------------------------------
__global__ void cvt_kernel(const float* __restrict__ src, bf16* __restrict__ dst, int n) {
  int i = (blockIdx.x * blockDim.x + threadIdx.x) * 4;
  if (i + 3 < n) {
    float4 f = *(const float4*)(src + i);
    dst[i + 0] = f2bf(f.x); dst[i + 1] = f2bf(f.y);
    dst[i + 2] = f2bf(f.z); dst[i + 3] = f2bf(f.w);
  } else {
    for (; i < n; ++i) dst[i] = f2bf(src[i]);
  }
}

// ---------------------------------------------------------------------------
// K1: QKV projection.  C[m,n] = x[m,:] . w[n,:] + bias[n]
// m = b*S+s (4096), n = h*64+d (1024).  Writes bf16 [B,H,S,D].
// Block: 128x128 tile, 8 waves as 2(m)x4(n); double-buffered async staging.
// ---------------------------------------------------------------------------
__global__ __launch_bounds__(256) void qkv_kernel(
    const bf16* __restrict__ xb,
    const bf16* __restrict__ wq, const bf16* __restrict__ wk, const bf16* __restrict__ wv,
    const float* __restrict__ bq, const float* __restrict__ bk, const float* __restrict__ bv,
    bf16* __restrict__ qo, bf16* __restrict__ ko, bf16* __restrict__ vo) {
  __shared__ bf16 As[2][128 * 40];
  __shared__ bf16 Bs[2][128 * 40];
  const int tid = threadIdx.x;
  const int wid = tid >> 5, lane = tid & 31, hi = lane >> 4, ll = lane & 15;
  const int m0 = blockIdx.x * 128, n0 = blockIdx.y * 128;
  const bf16*  W    = (blockIdx.z == 0) ? wq : (blockIdx.z == 1) ? wk : wv;
  const float* bias = (blockIdx.z == 0) ? bq : (blockIdx.z == 1) ? bk : bv;
  bf16*        out  = (blockIdx.z == 0) ? qo : (blockIdx.z == 1) ? ko : vo;
  const int wm = wid >> 2, wn = wid & 3;

  v8f acc[4][2];
  const v8f vzero = {0.f,0.f,0.f,0.f,0.f,0.f,0.f,0.f};
#pragma unroll
  for (int mi = 0; mi < 4; ++mi)
#pragma unroll
    for (int ni = 0; ni < 2; ++ni) acc[mi][ni] = vzero;

  // prologue: async-stage tile 0 (2 A-chunks + 2 B-chunks per lane)
#pragma unroll
  for (int i = tid; i < 512; i += 256) {
    int row = i >> 2, c4 = i & 3;
    async_ld_b128(As[0] + row * 40 + c4 * 8, xb + (size_t)(m0 + row) * CIN + c4 * 8);
    async_ld_b128(Bs[0] + row * 40 + c4 * 8, W  + (size_t)(n0 + row) * CIN + c4 * 8);
  }

  for (int kt = 0; kt < 32; ++kt) {
    const int cur = kt & 1;
    if (kt + 1 < 32) {
      const int k0n = (kt + 1) * 32;
#pragma unroll
      for (int i = tid; i < 512; i += 256) {
        int row = i >> 2, c4 = i & 3;
        async_ld_b128(As[cur ^ 1] + row * 40 + c4 * 8,
                      xb + (size_t)(m0 + row) * CIN + k0n + c4 * 8);
        async_ld_b128(Bs[cur ^ 1] + row * 40 + c4 * 8,
                      W + (size_t)(n0 + row) * CIN + k0n + c4 * 8);
      }
      WAIT_ASYNC(0x4);           // current tile resident; next tile in flight
    } else {
      WAIT_ASYNC(0x0);
    }
    __syncthreads();
    v16bf bfr[2];
#pragma unroll
    for (int ni = 0; ni < 2; ++ni) bfr[ni] = frag_b(Bs[cur], wn * 32 + ni * 16 + ll, 40, 0, hi);
#pragma unroll
    for (int mi = 0; mi < 4; ++mi) {
      v16bf a = frag_a(As[cur], wm * 64 + mi * 16 + ll, 40, 0, hi);
#pragma unroll
      for (int ni = 0; ni < 2; ++ni) acc[mi][ni] = wmma_bf16(a, bfr[ni], acc[mi][ni]);
    }
    __syncthreads();
  }

#pragma unroll
  for (int mi = 0; mi < 4; ++mi)
#pragma unroll
    for (int ni = 0; ni < 2; ++ni) {
      const int n = n0 + wn * 32 + ni * 16 + ll;
      const float bv_ = bias[n];
      const int h = n >> 6, d = n & 63;
#pragma unroll
      for (int r = 0; r < 8; ++r) {
        const int row = m0 + wm * 64 + mi * 16 + 8 * hi + r;
        const int bb = row >> 11, s = row & 2047;
        out[(((size_t)(bb * NH + h)) * S_LEN + s) * HD + d] = f2bf(acc[mi][ni][r] + bv_);
      }
    }
}

// ---------------------------------------------------------------------------
// K2: softmax row stats. For each (b,h,q): m = max_k(q.k/8), Z = sum exp(.-m).
// Block = 128 q-rows x all 2048 k; WMMA QK^T tiles + online shfl reductions.
// ks tiles double-buffered via async LDS DMA.
// ---------------------------------------------------------------------------
__global__ __launch_bounds__(256) void stats_kernel(
    const bf16* __restrict__ q, const bf16* __restrict__ k,
    float* __restrict__ rowmax, float* __restrict__ rowrcp) {
  __shared__ bf16 qs[128 * 72];
  __shared__ bf16 ks[2][128 * 72];
  const int tid = threadIdx.x;
  const int wid = tid >> 5, lane = tid & 31, hi = lane >> 4, ll = lane & 15;
  const int bh = blockIdx.y, q0 = blockIdx.x * 128;

#pragma unroll
  for (int i = tid; i < 1024; i += 256) {
    int row = i >> 3, c8 = i & 7;
    async_ld_b128(qs + row * 72 + c8 * 8,
                  q + ((size_t)bh * S_LEN + q0 + row) * HD + c8 * 8);
    async_ld_b128(ks[0] + row * 72 + c8 * 8,
                  k + ((size_t)bh * S_LEN + row) * HD + c8 * 8);
  }
  WAIT_ASYNC(0x0);
  __syncthreads();
  const v16bf a0 = frag_a(qs, wid * 16 + ll, 72, 0, hi);
  const v16bf a1 = frag_a(qs, wid * 16 + ll, 72, 32, hi);

  float m_run[8], z_run[8];
#pragma unroll
  for (int r = 0; r < 8; ++r) { m_run[r] = -1e30f; z_run[r] = 0.f; }

  for (int kt = 0; kt < 16; ++kt) {
    const int cur = kt & 1;
    if (kt + 1 < 16) {
#pragma unroll
      for (int i = tid; i < 1024; i += 256) {
        int row = i >> 3, c8 = i & 7;
        async_ld_b128(ks[cur ^ 1] + row * 72 + c8 * 8,
                      k + ((size_t)bh * S_LEN + (kt + 1) * 128 + row) * HD + c8 * 8);
      }
      WAIT_ASYNC(0x4);
    } else {
      WAIT_ASYNC(0x0);
    }
    __syncthreads();
    v8f accs[8];
#pragma unroll
    for (int j = 0; j < 8; ++j) {
      v8f a = {0.f,0.f,0.f,0.f,0.f,0.f,0.f,0.f};
      a = wmma_bf16(a0, frag_b(ks[cur], j * 16 + ll, 72, 0, hi), a);
      a = wmma_bf16(a1, frag_b(ks[cur], j * 16 + ll, 72, 32, hi), a);
      accs[j] = a;
    }
    // online max/sumexp per row (rows live in r; cols across 16 lanes x 8 tiles)
#pragma unroll
    for (int r = 0; r < 8; ++r) {
      float tmax = -1e30f;
#pragma unroll
      for (int j = 0; j < 8; ++j) tmax = fmaxf(tmax, accs[j][r] * 0.125f);
      tmax = fmaxf(tmax, __shfl_xor(tmax, 1, 32));
      tmax = fmaxf(tmax, __shfl_xor(tmax, 2, 32));
      tmax = fmaxf(tmax, __shfl_xor(tmax, 4, 32));
      tmax = fmaxf(tmax, __shfl_xor(tmax, 8, 32));
      const float nm = fmaxf(m_run[r], tmax);
      float ps = 0.f;
#pragma unroll
      for (int j = 0; j < 8; ++j) ps += __expf(accs[j][r] * 0.125f - nm);
      ps += __shfl_xor(ps, 1, 32);
      ps += __shfl_xor(ps, 2, 32);
      ps += __shfl_xor(ps, 4, 32);
      ps += __shfl_xor(ps, 8, 32);
      z_run[r] = z_run[r] * __expf(m_run[r] - nm) + ps;
      m_run[r] = nm;
    }
    __syncthreads();
  }
  if (ll == 0) {
#pragma unroll
    for (int r = 0; r < 8; ++r) {
      const int qq = q0 + wid * 16 + 8 * hi + r;
      rowmax[(size_t)bh * S_LEN + qq] = m_run[r];
      rowrcp[(size_t)bh * S_LEN + qq] = 1.0f / z_run[r];
    }
  }
}

// ---------------------------------------------------------------------------
// K2b: colsum(v) per (b,h,d)
// ---------------------------------------------------------------------------
__global__ void vcolsum_kernel(const bf16* __restrict__ v, float* __restrict__ vcs) {
  const int bh = blockIdx.x, d = threadIdx.x;  // 64 threads
  float s = 0.f;
  for (int i = 0; i < S_LEN; ++i) s += bf2f(v[((size_t)bh * S_LEN + i) * HD + d]);
  vcs[bh * HD + d] = s;
}

// ---------------------------------------------------------------------------
// K3: t = attn^T @ v  per (b,h).  Block owns 128 k-rows of t, loops q in 32s:
//   phase1: s^T[k,q] = K . Q^T via WMMA, exp-normalize -> attnT (LDS bf16)
//   phase2: t[k,d]  += attnT @ v via WMMA (v staged transposed as [d][q])
// kb + qs staged via async LDS DMA; vT transpose stays VGPR-mediated.
// ---------------------------------------------------------------------------
__global__ __launch_bounds__(256) void tmat_kernel(
    const bf16* __restrict__ q, const bf16* __restrict__ k, const bf16* __restrict__ v,
    const float* __restrict__ rowmax, const float* __restrict__ rowrcp,
    bf16* __restrict__ t) {
  __shared__ bf16 kb[128 * 72];
  __shared__ bf16 qs[32 * 72];
  __shared__ bf16 attnT[128 * 40];
  __shared__ bf16 vT[64 * 40];
  __shared__ float statm[32];
  __shared__ float statr[32];
  const int tid = threadIdx.x;
  const int wid = tid >> 5, lane = tid & 31, hi = lane >> 4, ll = lane & 15;
  const int bh = blockIdx.y, k0 = blockIdx.x * 128;

#pragma unroll
  for (int i = tid; i < 1024; i += 256) {
    int row = i >> 3, c8 = i & 7;
    async_ld_b128(kb + row * 72 + c8 * 8,
                  k + ((size_t)bh * S_LEN + k0 + row) * HD + c8 * 8);
  }
  WAIT_ASYNC(0x0);
  __syncthreads();
  const v16bf ak0 = frag_a(kb, wid * 16 + ll, 72, 0, hi);
  const v16bf ak1 = frag_a(kb, wid * 16 + ll, 72, 32, hi);

  v8f accT[4];
  const v8f vzero = {0.f,0.f,0.f,0.f,0.f,0.f,0.f,0.f};
#pragma unroll
  for (int j = 0; j < 4; ++j) accT[j] = vzero;

  for (int qc = 0; qc < 64; ++qc) {
    const int q0 = qc * 32;
    {  // stage q chunk [32][64] (async), stats + v chunk transposed [64][32] (sync)
      int row = tid >> 3, c8 = tid & 7;
      async_ld_b128(qs + row * 72 + c8 * 8,
                    q + ((size_t)bh * S_LEN + q0 + row) * HD + c8 * 8);
      union { uint4 u; bf16 h[8]; } tv;
      tv.u = *(const uint4*)(v + ((size_t)bh * S_LEN + q0 + row) * HD + c8 * 8);
#pragma unroll
      for (int e = 0; e < 8; ++e) vT[(c8 * 8 + e) * 40 + row] = tv.h[e];
      if (tid < 32) {
        statm[tid] = rowmax[(size_t)bh * S_LEN + q0 + tid];
        statr[tid] = rowrcp[(size_t)bh * S_LEN + q0 + tid];
      }
    }
    WAIT_ASYNC(0x0);
    __syncthreads();
    // phase 1: scores^T and normalized attn^T into LDS
#pragma unroll
    for (int j = 0; j < 2; ++j) {
      v8f s = vzero;
      s = wmma_bf16(ak0, frag_b(qs, j * 16 + ll, 72, 0, hi), s);
      s = wmma_bf16(ak1, frag_b(qs, j * 16 + ll, 72, 32, hi), s);
      const int qcol = j * 16 + ll;
      const float m = statm[qcol], rz = statr[qcol];
#pragma unroll
      for (int r = 0; r < 8; ++r) {
        const int krow = wid * 16 + 8 * hi + r;
        attnT[krow * 40 + qcol] = f2bf(__expf(s[r] * 0.125f - m) * rz);
      }
    }
    __syncthreads();
    // phase 2: t += attnT(16k x 32q) @ v(32q x 64d)
    const v16bf a = frag_a(attnT, wid * 16 + ll, 40, 0, hi);
#pragma unroll
    for (int j = 0; j < 4; ++j)
      accT[j] = wmma_bf16(a, frag_b(vT, j * 16 + ll, 40, 0, hi), accT[j]);
    __syncthreads();
  }

#pragma unroll
  for (int j = 0; j < 4; ++j) {
    const int d = j * 16 + ll;
#pragma unroll
    for (int r = 0; r < 8; ++r) {
      const int kr = k0 + wid * 16 + 8 * hi + r;
      t[((size_t)bh * S_LEN + kr) * HD + d] = f2bf(accT[j][r]);
    }
  }
}

// ---------------------------------------------------------------------------
// K4: out[b,o,h*64+d] = wf[o,:] . t[bh,:,d] + bf[o]*colsum_v[bh,d]   (fp32 out)
// Block: 128(o) x 64(d) tile; waves 4(m)x2(n).
// wf operand (8MB stream) double-buffered via async LDS DMA.
// ---------------------------------------------------------------------------
__global__ __launch_bounds__(256) void out_kernel(
    const bf16* __restrict__ wfb, const bf16* __restrict__ t,
    const float* __restrict__ bf_, const float* __restrict__ vcs,
    float* __restrict__ out) {
  __shared__ bf16 As[2][128 * 40];
  __shared__ bf16 Bs[64 * 40];
  const int tid = threadIdx.x;
  const int wid = tid >> 5, lane = tid & 31, hi = lane >> 4, ll = lane & 15;
  const int bh = blockIdx.y, o0 = blockIdx.x * 128;
  const int b = bh >> 4, h = bh & 15;
  const int wm = wid >> 1, wn = wid & 1;

  v8f acc[2][2];
  const v8f vzero = {0.f,0.f,0.f,0.f,0.f,0.f,0.f,0.f};
#pragma unroll
  for (int mi = 0; mi < 2; ++mi)
#pragma unroll
    for (int ni = 0; ni < 2; ++ni) acc[mi][ni] = vzero;

  // prologue: async-stage wf tile 0 (2 chunks per lane)
#pragma unroll
  for (int i = tid; i < 512; i += 256) {
    int row = i >> 2, c4 = i & 3;
    async_ld_b128(As[0] + row * 40 + c4 * 8, wfb + (size_t)(o0 + row) * S_LEN + c4 * 8);
  }

  for (int kt = 0; kt < 64; ++kt) {
    const int cur = kt & 1;
    const int k0 = kt * 32;
    if (kt + 1 < 64) {
      const int k0n = k0 + 32;
#pragma unroll
      for (int i = tid; i < 512; i += 256) {
        int row = i >> 2, c4 = i & 3;
        async_ld_b128(As[cur ^ 1] + row * 40 + c4 * 8,
                      wfb + (size_t)(o0 + row) * S_LEN + k0n + c4 * 8);
      }
      WAIT_ASYNC(0x2);
    } else {
      WAIT_ASYNC(0x0);
    }
    {  // stage t tile transposed: [32 k][64 d] -> Bs[d][k]  (sync, VGPR transpose)
      int krow = tid >> 3, c8 = tid & 7;
      union { uint4 u; bf16 h[8]; } tv;
      tv.u = *(const uint4*)(t + ((size_t)bh * S_LEN + k0 + krow) * HD + c8 * 8);
#pragma unroll
      for (int e = 0; e < 8; ++e) Bs[(c8 * 8 + e) * 40 + krow] = tv.h[e];
    }
    __syncthreads();
    v16bf bfr[2];
#pragma unroll
    for (int ni = 0; ni < 2; ++ni) bfr[ni] = frag_b(Bs, wn * 32 + ni * 16 + ll, 40, 0, hi);
#pragma unroll
    for (int mi = 0; mi < 2; ++mi) {
      v16bf a = frag_a(As[cur], wm * 32 + mi * 16 + ll, 40, 0, hi);
#pragma unroll
      for (int ni = 0; ni < 2; ++ni) acc[mi][ni] = wmma_bf16(a, bfr[ni], acc[mi][ni]);
    }
    __syncthreads();
  }

#pragma unroll
  for (int mi = 0; mi < 2; ++mi)
#pragma unroll
    for (int ni = 0; ni < 2; ++ni) {
      const int d = wn * 32 + ni * 16 + ll;
      const float cs = vcs[bh * HD + d];
#pragma unroll
      for (int r = 0; r < 8; ++r) {
        const int o = o0 + wm * 32 + mi * 16 + 8 * hi + r;
        out[((size_t)(b * S_LEN + o)) * CIN + h * HD + d] = acc[mi][ni][r] + bf_[o] * cs;
      }
    }
}

// ---------------------------------------------------------------------------
// Host launcher
// ---------------------------------------------------------------------------
extern "C" void kernel_launch(void* const* d_in, const int* in_sizes, int n_in,
                              void* d_out, int out_size, void* d_ws, size_t ws_size,
                              hipStream_t stream) {
  (void)in_sizes; (void)n_in; (void)out_size; (void)ws_size;
  const float* x  = (const float*)d_in[0];
  const float* wq = (const float*)d_in[1];
  const float* bq = (const float*)d_in[2];
  const float* wk = (const float*)d_in[3];
  const float* bk = (const float*)d_in[4];
  const float* wv = (const float*)d_in[5];
  const float* bv = (const float*)d_in[6];
  const float* wf = (const float*)d_in[7];
  const float* bfv = (const float*)d_in[8];
  float* out = (float*)d_out;

  char* ws = (char*)d_ws;
  size_t off = 0;
  bf16* xb  = (bf16*)(ws + off); off += (size_t)4096 * 1024 * 2;   // x bf16
  bf16* wqb = (bf16*)(ws + off); off += (size_t)1024 * 1024 * 2;
  bf16* wkb = (bf16*)(ws + off); off += (size_t)1024 * 1024 * 2;
  bf16* wvb = (bf16*)(ws + off); off += (size_t)1024 * 1024 * 2;
  bf16* wfb = (bf16*)(ws + off); off += (size_t)2048 * 2048 * 2;
  bf16* qb  = (bf16*)(ws + off); off += (size_t)BHN * S_LEN * HD * 2;
  bf16* kb  = (bf16*)(ws + off); off += (size_t)BHN * S_LEN * HD * 2;
  bf16* vb  = (bf16*)(ws + off); off += (size_t)BHN * S_LEN * HD * 2;
  bf16* tb  = (bf16*)(ws + off); off += (size_t)BHN * S_LEN * HD * 2;
  float* rowmax = (float*)(ws + off); off += (size_t)BHN * S_LEN * 4;
  float* rowrcp = (float*)(ws + off); off += (size_t)BHN * S_LEN * 4;
  float* vcs    = (float*)(ws + off); off += (size_t)BHN * HD * 4;

  // K0: fp32 -> bf16
  const int XN = 4096 * 1024, WN = 1024 * 1024, FN = 2048 * 2048;
  cvt_kernel<<<(XN / 4 + 255) / 256, 256, 0, stream>>>(x,  xb,  XN);
  cvt_kernel<<<(WN / 4 + 255) / 256, 256, 0, stream>>>(wq, wqb, WN);
  cvt_kernel<<<(WN / 4 + 255) / 256, 256, 0, stream>>>(wk, wkb, WN);
  cvt_kernel<<<(WN / 4 + 255) / 256, 256, 0, stream>>>(wv, wvb, WN);
  cvt_kernel<<<(FN / 4 + 255) / 256, 256, 0, stream>>>(wf, wfb, FN);

  // K1: QKV projections (32 m-tiles x 8 n-tiles x {q,k,v})
  qkv_kernel<<<dim3(32, 8, 3), 256, 0, stream>>>(xb, wqb, wkb, wvb, bq, bk, bv, qb, kb, vb);

  // K2: softmax row stats + colsum(v)
  stats_kernel<<<dim3(16, BHN), 256, 0, stream>>>(qb, kb, rowmax, rowrcp);
  vcolsum_kernel<<<BHN, 64, 0, stream>>>(vb, vcs);

  // K3: t = attn^T @ v
  tmat_kernel<<<dim3(16, BHN), 256, 0, stream>>>(qb, kb, vb, rowmax, rowrcp, tb);

  // K4: out = wf @ t + bf * colsum(v)
  out_kernel<<<dim3(16, BHN), 256, 0, stream>>>(wfb, tb, bfv, vcs, out);
}
